// HawkesDecayRNN_22780506538516
// MI455X (gfx1250) — matrix operand order repaired and checked by reading, not compiled, not run on hardware
//
#include <hip/hip_runtime.h>
#include <cstdint>

#define S_LEN 2048
#define B_SZ  256
#define K_DIM 64
#define H_DIM 128
#define CCOL  (K_DIM + H_DIM)   // 192 concat columns
#define CSTR  200               // padded concat row stride in bf16 elems (400B, 16B aligned)
#define DSTR  132               // padded decay row stride in f32

typedef __attribute__((ext_vector_type(16))) __bf16 v16bf;
typedef __attribute__((ext_vector_type(8)))  float  v8f;

union FragA { struct { uint4 lo, hi; } q; v16bf v; };
union FragB { unsigned short us[16]; v16bf v; };

static __device__ inline unsigned short f2bf(float f) {
  union { float f; unsigned u; } c; c.f = f;
  unsigned u = c.u;
  u = u + 0x7FFFu + ((u >> 16) & 1u);   // round-to-nearest-even
  return (unsigned short)(u >> 16);
}

// Branch-free fast transcendentals (v_exp_f32 / v_log_f32 / v_rcp_f32),
// always numerically stable forms -> no divergent libm slow paths.
static __device__ inline float fast_tanh(float x) {
  const float e = __expf(-2.0f * __builtin_fabsf(x));       // in (0,1]
  const float t = (1.0f - e) * __builtin_amdgcn_rcpf(1.0f + e);
  return __builtin_copysignf(t, x);
}

// softplus(beta=10): (1/10)*log1p(exp(10 v)) = max(v,0) + 0.1*log(1+exp(-10|v|))
static __device__ inline float softplus10(float v) {
  const float e = __expf(-10.0f * __builtin_fabsf(v));      // in (0,1]
  return __builtin_fmaxf(v, 0.0f) + 0.1f * __logf(1.0f + e);
}

// One workgroup = 16 waves = one 16-row batch tile, persistent over all S steps.
// Waves 0..7  : decay GEMM tiles (N-tile = wave&7), softplus10 epilogue
// Waves 8..15 : hidden GEMM tiles, tanh + exp-decay epilogue
// Concat (x|h) and dt are double-buffered in LDS -> 2 barriers per step.
// Input prefetch is pipelined across iterations with the seq->embed dependent
// chain split over two steps: during step t we issue embed(row seq(t+1)),
// seq(t+2), dt(t+1); their only wait lands at the publish point of step t+1,
// a full step after issue -> no global-load latency on the recurrence chain.
__global__ __launch_bounds__(512, 1)
void hawkes_decay_rnn_kernel(const float* __restrict__ dt,
                             const float* __restrict__ h0,
                             const float* __restrict__ embed_W,
                             const float* __restrict__ W_ih,
                             const float* __restrict__ b_ih,
                             const float* __restrict__ W_hh,
                             const float* __restrict__ b_hh,
                             const float* __restrict__ dec_W,
                             const float* __restrict__ dec_b,
                             const int*   __restrict__ seq_types,
                             float* __restrict__ out) {
  __shared__ __align__(16) unsigned short lds_concat[2][16 * CSTR]; // [x(64)|h(128)] bf16
  __shared__ float lds_decay[16 * DSTR];
  __shared__ float lds_dt[2][16];

  const int tid   = threadIdx.x;
  const int lane  = tid & 31;
  const int wave  = tid >> 5;
  const int mat   = wave >> 3;        // 0 = decay GEMM, 1 = hidden GEMM (wave-uniform)
  const int nt    = wave & 7;         // N tile within H
  const int ncol  = lane & 15;        // N within tile (C/D layout: N = lane&15)
  const int hh    = lane >> 4;        // lane half
  const int nglob = nt * 16 + ncol;   // column in H
  const int rowA  = lane & 15;        // A-matrix row carried by this lane
  const int b0    = blockIdx.x * 16;  // batch tile origin
  const int r     = tid >> 5;         // gather row this thread fills
  const int c0    = (tid & 31) * 2;   // gather column pair this thread fills

  float* const outH = out;                                        // hiddens
  float* const outD = out + (size_t)S_LEN * B_SZ * H_DIM;         // decays
  float* const outT = out + (size_t)2 * S_LEN * B_SZ * H_DIM;     // hiddens_ti

  // ---- preload per-wave constant weight (B-operand) fragments, f32 -> bf16 ----
  // B layout (32x16 bf16): lane half hh covers K = 16*hh + j, j = 0..15; N = lane&15.
  FragB bfrag[6];
  #pragma unroll
  for (int kt = 0; kt < 6; ++kt) {
    const int kbase = kt * 32 + 16 * hh;
    const float* src;
    if (mat == 0)            src = dec_W + (size_t)nglob * CCOL + kbase;
    else if (kbase < K_DIM)  src = W_ih  + (size_t)nglob * K_DIM + kbase;
    else                     src = W_hh  + (size_t)nglob * H_DIM + (kbase - K_DIM);
    #pragma unroll
    for (int j = 0; j < 16; ++j) bfrag[kt].us[j] = f2bf(src[j]);
  }
  const float bias = (mat == 0) ? dec_b[nglob] : (b_ih[nglob] + b_hh[nglob]);

  // ---- init resident h-region of concat buffer 0 from h0 ----
  {
    const int c4 = (tid & 31) * 4;
    #pragma unroll
    for (int j = 0; j < 4; ++j)
      lds_concat[0][r * CSTR + K_DIM + c4 + j] =
          f2bf(h0[(size_t)(b0 + r) * H_DIM + c4 + j]);
  }

  // ---- prologue of the input pipeline ----
  // xe   : embed pair for step 0
  // dtv  : dt for step 0
  // ty_nx: seq type for step 1 (embed row issued during step 0)
  const int ty0 = seq_types[b0 + r];
  int ty_nx     = seq_types[B_SZ + b0 + r];
  float2 xe     = *(const float2*)(embed_W + (size_t)ty0 * K_DIM + c0);
  float  dtv    = dt[b0 + (tid & 15)];

  int cur = 0;
  #pragma unroll 1
  for (int t = 0; t < S_LEN; ++t) {
    unsigned short* const C  = lds_concat[cur];      // read this step
    unsigned short* const Cn = lds_concat[cur ^ 1];  // written for next step

    // ---- publish this step's prefetched inputs (waits drained here, a full
    //      step after the loads were issued) ----
    C[r * CSTR + c0 + 0] = f2bf(xe.x);
    C[r * CSTR + c0 + 1] = f2bf(xe.y);
    if (tid < 16) lds_dt[cur][tid] = dtv;
    const int ty = ty_nx;  // seq(t+1): value already resident (loaded last step)
    __syncthreads();  // barrier A: concat(t) + dt(t) ready; prev-step reads done

    // ---- issue prefetches for the next steps (no dependent waits in-step) ----
    {
      const int t1 = (t + 1 < S_LEN) ? (t + 1) : t;
      const int t2 = (t + 2 < S_LEN) ? (t + 2) : t;
      xe    = *(const float2*)(embed_W + (size_t)ty * K_DIM + c0); // x(t+1)
      ty_nx = seq_types[(size_t)t2 * B_SZ + b0 + r];               // seq(t+2)
      dtv   = dt[(size_t)t1 * B_SZ + b0 + (tid & 15)];             // dt(t+1)
    }

    // ---- acc(16x16) = concat_tile(16x192) x W^T tile, 6-WMMA K chain ----
    // A layout (16x32 bf16): lane half hh holds K = kt*32 + 16*hh + {0..15}.
    FragA a[6];
    #pragma unroll
    for (int kt = 0; kt < 6; ++kt) {
      const unsigned short* base = &C[rowA * CSTR + kt * 32 + 16 * hh];
      a[kt].q.lo = *(const uint4*)(base);        // K = kt*32 + 16*hh + 0..7
      a[kt].q.hi = *(const uint4*)(base + 8);    // K = kt*32 + 16*hh + 8..15
    }
    v8f acc = {};
    #pragma unroll
    for (int kt = 0; kt < 6; ++kt)
      acc = __builtin_amdgcn_wmma_f32_16x16x32_bf16(
          false, a[kt].v, false, bfrag[kt].v, (short)0, acc, false, false);

    // ---- epilogue phase 1 (wave-uniform branch, barriers stay convergent) ----
    float hid[8];
    if (mat == 0) {
      #pragma unroll
      for (int i = 0; i < 8; ++i) {             // C/D layout: M = 8*hh + i
        const int m = 8 * hh + i;
        const float sp = softplus10(acc[i] + bias);
        lds_decay[m * DSTR + nglob] = sp;
        outD[((size_t)t * B_SZ + (b0 + m)) * H_DIM + nglob] = sp;
      }
    } else {
      #pragma unroll
      for (int i = 0; i < 8; ++i) {
        const int m = 8 * hh + i;
        hid[i] = fast_tanh(acc[i] + bias);
        outH[((size_t)t * B_SZ + (b0 + m)) * H_DIM + nglob] = hid[i];
      }
    }
    __syncthreads();  // barrier B: decay published

    // ---- epilogue phase 2: h_new = hidden * exp(-decay*dt) -> next buffer ----
    if (mat == 1) {
      #pragma unroll
      for (int i = 0; i < 8; ++i) {
        const int m = 8 * hh + i;
        const float dec = lds_decay[m * DSTR + nglob];
        const float hn  = hid[i] * __expf(-dec * lds_dt[cur][m]);
        Cn[m * CSTR + K_DIM + nglob] = f2bf(hn);  // h(t) for next step
        outT[((size_t)t * B_SZ + (b0 + m)) * H_DIM + nglob] = hn;
      }
    }
    cur ^= 1;
    // next iteration's barrier A orders Cn writes vs next-step reads
  }
}

extern "C" void kernel_launch(void* const* d_in, const int* in_sizes, int n_in,
                              void* d_out, int out_size, void* d_ws, size_t ws_size,
                              hipStream_t stream) {
  (void)in_sizes; (void)n_in; (void)out_size; (void)d_ws; (void)ws_size;
  hawkes_decay_rnn_kernel<<<dim3(B_SZ / 16), dim3(512), 0, stream>>>(
      (const float*)d_in[0],  // dt        [S,B]
      (const float*)d_in[1],  // h0        [B,H]
      (const float*)d_in[2],  // embed_W   [K+1,K]
      (const float*)d_in[3],  // W_ih      [H,K]
      (const float*)d_in[4],  // b_ih      [H]
      (const float*)d_in[5],  // W_hh      [H,H]
      (const float*)d_in[6],  // b_hh      [H]
      (const float*)d_in[7],  // dec_W     [H,K+H]
      (const float*)d_in[8],  // dec_b     [H]
      (const int*)  d_in[9],  // seq_types [S,B]
      (float*)d_out);         // [hiddens | decays | hiddens_ti]
}
